// Qwen3NextGatedDeltaNet_17867063951956
// MI455X (gfx1250) — matrix operand order repaired and checked by reading, not compiled, not run on hardware
//
#include <hip/hip_runtime.h>
#include <math.h>

// ---------------- types ----------------
typedef __bf16 bf16;
typedef __bf16 v16bf  __attribute__((ext_vector_type(16)));
typedef __bf16 bf16x8 __attribute__((ext_vector_type(8)));
typedef float  v8f    __attribute__((ext_vector_type(8)));

// ---------------- problem constants ----------------
#define B_    2
#define S_    2048
#define D_    2048
#define HQK_  16
#define HV_   32
#define HD_   128
#define CHUNK_ 64
#define NC_   (S_/CHUNK_)          // 32
#define BS_   (B_*S_)              // 4096
#define OQ_   ((2*HQK_+2*HV_)*HD_) // 12288
#define OBA_  (2*HV_)              // 64
#define CD_   ((2*HQK_+HV_)*HD_)   // 8192

// ---------------- WMMA helpers ----------------
__device__ __forceinline__ v8f wmma_bf16(v16bf a, v16bf b, v8f c) {
  // D = A(16x32 bf16) x B(32x16 bf16) + C(16x16 f32)
  return __builtin_amdgcn_wmma_f32_16x16x32_bf16(false, a, false, b, (short)0, c, false, false);
}
__device__ __forceinline__ v8f zero8() {
  v8f c;
#pragma unroll
  for (int q = 0; q < 8; q++) c[q] = 0.f;
  return c;
}
__device__ __forceinline__ v16bf frag_cat(const bf16* p0, const bf16* p1) {
  bf16x8 lo = *(const bf16x8*)p0;
  bf16x8 hi = *(const bf16x8*)p1;
  return __builtin_shufflevector(lo, hi, 0,1,2,3,4,5,6,7,8,9,10,11,12,13,14,15);
}
// A fragment (16x32): row-major buffer, lane L holds row m0+(L&15),
// K = (L>>4)*8 + [0..7]  and  16+(L>>4)*8 + [0..7]   (ISA 7.12.2)
__device__ __forceinline__ v16bf fragA(const bf16* base, int stride, int m0, int k0, int lane) {
  const bf16* p = base + (m0 + (lane & 15)) * stride + k0 + ((lane >> 4) << 3);
  return frag_cat(p, p + 16);
}
// B fragment (32x16): buffer holds B^T row-major (rows = N, k contiguous),
// lane L holds col n0+(L&15), K = (L>>4)*8 + [0..7]  and  16+(L>>4)*8 + [0..7]
__device__ __forceinline__ v16bf fragB(const bf16* base, int stride, int n0, int k0, int lane) {
  const bf16* p = base + (n0 + (lane & 15)) * stride + k0 + ((lane >> 4) << 3);
  return frag_cat(p, p + 16);
}

// ---------------- cast kernel ----------------
__global__ void cast_f32_bf16(const float* __restrict__ src, bf16* __restrict__ dst, long n) {
  long i = (long)blockIdx.x * blockDim.x + threadIdx.x;
  long gs = (long)gridDim.x * blockDim.x;
  for (; i < n; i += gs) dst[i] = (bf16)src[i];
}

// ---------------- WMMA GEMM: C[M,N] = A[M,K] * Bt[N,K]^T ----------------
// Block tile 128x128, 8 waves (4 along M x 2 along N), each wave 32x64 = 8 wmma tiles.
#define BK_  32
#define LDA_ 40   // padded LDS row stride (halves), 80B -> 16B aligned chunks
__global__ void __launch_bounds__(256) gemm_nt_bf16(
    const bf16* __restrict__ A, const bf16* __restrict__ Bt,
    float* __restrict__ C, int M, int N, int K) {
  __shared__ bf16 As[128][LDA_];
  __shared__ bf16 Bs[128][LDA_];
  const int tid = threadIdx.x, lane = tid & 31, wid = tid >> 5;
  const int wm = (wid & 3) * 32, wn = (wid >> 2) * 64;
  const int bm = blockIdx.y * 128, bn = blockIdx.x * 128;
  v8f acc[2][4];
#pragma unroll
  for (int i = 0; i < 2; i++)
#pragma unroll
    for (int j = 0; j < 4; j++) acc[i][j] = zero8();

  const int lrow = tid >> 1, lcol = (tid & 1) * 16;
  for (int k0 = 0; k0 < K; k0 += BK_) {
    { // A tile: 128 x 32
      const bf16* p = A + (long)(bm + lrow) * K + k0 + lcol;
      *(bf16x8*)&As[lrow][lcol]     = *(const bf16x8*)p;
      *(bf16x8*)&As[lrow][lcol + 8] = *(const bf16x8*)(p + 8);
      if (k0 + BK_ < K) __builtin_prefetch(p + BK_, 0, 0);
    }
    { // B tile: 128 x 32 (rows of Bt), guard N
      const int nrow = bn + lrow;
      if (nrow < N) {
        const bf16* p = Bt + (long)nrow * K + k0 + lcol;
        *(bf16x8*)&Bs[lrow][lcol]     = *(const bf16x8*)p;
        *(bf16x8*)&Bs[lrow][lcol + 8] = *(const bf16x8*)(p + 8);
        if (k0 + BK_ < K) __builtin_prefetch(p + BK_, 0, 0);
      } else {
        bf16x8 z;
#pragma unroll
        for (int q = 0; q < 8; q++) z[q] = (bf16)0.f;
        *(bf16x8*)&Bs[lrow][lcol]     = z;
        *(bf16x8*)&Bs[lrow][lcol + 8] = z;
      }
    }
    __syncthreads();
    v16bf a0 = fragA(&As[0][0], LDA_, wm,      0, lane);
    v16bf a1 = fragA(&As[0][0], LDA_, wm + 16, 0, lane);
#pragma unroll
    for (int nt = 0; nt < 4; nt++) {
      v16bf b = fragB(&Bs[0][0], LDA_, wn + nt * 16, 0, lane);
      acc[0][nt] = wmma_bf16(a0, b, acc[0][nt]);
      acc[1][nt] = wmma_bf16(a1, b, acc[1][nt]);
    }
    __syncthreads();
  }
#pragma unroll
  for (int mt = 0; mt < 2; mt++)
#pragma unroll
    for (int nt = 0; nt < 4; nt++) {
      int m0 = bm + wm + mt * 16 + ((lane >> 4) << 3);
      int n0 = bn + wn + nt * 16 + (lane & 15);
      if (n0 < N) {
#pragma unroll
        for (int r = 0; r < 8; r++) C[(long)(m0 + r) * N + n0] = acc[mt][nt][r];
      }
    }
}

// ---------------- depthwise causal conv(K=4) + SiLU ----------------
__global__ void conv_silu_kernel(const float* __restrict__ mqkvz,
                                 const float* __restrict__ convw,
                                 float* __restrict__ qconv, float* __restrict__ kconv,
                                 bf16* __restrict__ vconv) {
  const int b = blockIdx.y;
  const int c = blockIdx.x * blockDim.x + threadIdx.x;
  if (c >= CD_) return;
  int col, kind, dc;
  if (c < 2048)      { int h = c >> 7;        col = h * 768 + (c & 127);                       kind = 0; dc = c; }
  else if (c < 4096) { int cc = c - 2048;     col = (cc >> 7) * 768 + 128 + (cc & 127);        kind = 1; dc = cc; }
  else               { int cc = c - 4096; int vh = cc >> 7;
                       col = (vh >> 1) * 768 + 256 + (vh & 1) * 128 + (cc & 127);              kind = 2; dc = cc; }
  const float w0 = convw[c*4+0], w1 = convw[c*4+1], w2 = convw[c*4+2], w3 = convw[c*4+3];
  float x1 = 0.f, x2 = 0.f, x3 = 0.f;
  const float* src = mqkvz + (long)b * S_ * OQ_ + col;
  for (int t = 0; t < S_; t++) {
    float x0 = src[(long)t * OQ_];
    float y = w0*x3 + w1*x2 + w2*x1 + w3*x0;
    y = y / (1.f + expf(-y));                 // SiLU
    long o = (long)(b * S_ + t);
    if (kind == 0)      qconv[o * 2048 + dc] = y;
    else if (kind == 1) kconv[o * 2048 + dc] = y;
    else                vconv[o * 4096 + dc] = (bf16)y;
    x3 = x2; x2 = x1; x1 = x0;
  }
}

// ---------------- beta / g ----------------
__global__ void gate_kernel(const float* __restrict__ mba, const float* __restrict__ A_log,
                            const float* __restrict__ dt_bias,
                            float* __restrict__ gbuf, float* __restrict__ bbuf) {
  int i = blockIdx.x * blockDim.x + threadIdx.x;
  if (i >= BS_ * HV_) return;
  int hv = i & 31; long bs = i >> 5;
  int h = hv >> 1, j = hv & 1;
  float braw = mba[bs * OBA_ + h * 4 + j];
  float araw = mba[bs * OBA_ + h * 4 + 2 + j];
  bbuf[i] = 1.f / (1.f + expf(-braw));
  float xa = araw + dt_bias[hv];
  float sp = (xa > 20.f) ? xa : log1pf(expf(xa));
  gbuf[i] = -expf(A_log[hv]) * sp;
}

// ---------------- per-row RMSNorm (128 wide) -> bf16 ----------------
__global__ void __launch_bounds__(256) rmsnorm_rows(const float* __restrict__ src,
                                                    bf16* __restrict__ dst,
                                                    float scale, int nrows) {
  int lane = threadIdx.x & 31, wid = threadIdx.x >> 5;
  int row = blockIdx.x * 8 + wid;
  if (row >= nrows) return;
  const float* p = src + (long)row * HD_;
  float v[4]; float ss = 0.f;
#pragma unroll
  for (int j = 0; j < 4; j++) { v[j] = p[lane + 32*j]; ss += v[j]*v[j]; }
#pragma unroll
  for (int m = 16; m > 0; m >>= 1) ss += __shfl_xor(ss, m, 32);
  float r = rsqrtf(ss * (1.f / HD_) + 1e-6f) * scale;
  bf16* q = dst + (long)row * HD_;
#pragma unroll
  for (int j = 0; j < 4; j++) q[lane + 32*j] = (bf16)(v[j] * r);
}

// ---------------- chunk precompute: cumg, T=(I+M)^-1, v_c, k_cumdecay ----------------
__global__ void __launch_bounds__(256) chunk_precompute(
    const bf16* __restrict__ kn_h, const bf16* __restrict__ vconv,
    const float* __restrict__ gbuf, const float* __restrict__ bbuf,
    float* __restrict__ cumg_g, bf16* __restrict__ vc_h, bf16* __restrict__ kcd_h) {
  __shared__ bf16  knS[64][136];
  __shared__ bf16  kbS[64][136];
  __shared__ float Mm[64][65];
  __shared__ float Tm[64][65];
  __shared__ bf16  Tb[64][72];
  __shared__ bf16  vbT[128][72];
  __shared__ bf16  kgT[128][72];
  __shared__ float cg[64];
  __shared__ float bet[64];

  const int tid = threadIdx.x, lane = tid & 31, wid = tid >> 5;
  const int blk = blockIdx.x;             // ((b*HV+hv)*NC + ic)
  const int ic = blk & 31, bh = blk >> 5;
  const int hv = bh & 31,  b  = bh >> 5;
  const int h = hv >> 1;
  const int s0 = ic * CHUNK_;

  if (tid == 0) {
    float acc = 0.f;
    for (int t = 0; t < 64; t++) {
      acc += gbuf[(long)(b * S_ + s0 + t) * HV_ + hv];
      cg[t] = acc;
    }
  }
  if (tid < 64) bet[tid] = bbuf[(long)(b * S_ + s0 + tid) * HV_ + hv];
  __syncthreads();
  if (tid < 64) cumg_g[((long)(b * HV_ + hv) * NC_ + ic) * 64 + tid] = cg[tid];

  for (int idx = tid; idx < 64 * 128; idx += 256) {
    int t = idx >> 7, d = idx & 127;
    float kv = (float)kn_h[((long)(b * S_ + s0 + t) * HQK_ + h) * HD_ + d];
    float kb = kv * bet[t];
    knS[t][d] = (bf16)kv;
    kbS[t][d] = (bf16)kb;
    kgT[d][t] = (bf16)(kb * expf(cg[t]));
    float vv = (float)vconv[((long)(b * S_ + s0 + t) * HV_ + hv) * HD_ + d];
    vbT[d][t] = (bf16)(vv * bet[t]);
  }
  __syncthreads();

  // M[i][j] = (kb . kn) * exp(cg_i - cg_j), strictly lower
  for (int tt = wid; tt < 16; tt += 8) {
    int mt = tt >> 2, nt = tt & 3;
    v8f c = zero8();
    for (int kk = 0; kk < 128; kk += 32) {
      v16bf a  = fragA(&kbS[0][0], 136, mt * 16, kk, lane);
      v16bf bb = fragB(&knS[0][0], 136, nt * 16, kk, lane);
      c = wmma_bf16(a, bb, c);
    }
#pragma unroll
    for (int r = 0; r < 8; r++) {
      int i = mt * 16 + r + ((lane >> 4) << 3);
      int j = nt * 16 + (lane & 15);
      Mm[i][j] = (j < i) ? c[r] * expf(cg[i] - cg[j]) : 0.f;
    }
  }
  __syncthreads();

  // forward substitution: T = (I+M)^-1 (unit lower); thread `c` owns column c
  if (tid < 64) {
    int ccol = tid;
    for (int i = 0; i < 64; i++) {
      float acc2 = (i == ccol) ? 1.f : 0.f;
      for (int j = ccol; j < i; j++) acc2 -= Mm[i][j] * Tm[j][ccol];
      Tm[i][ccol] = acc2;
    }
  }
  __syncthreads();
  for (int idx = tid; idx < 64 * 64; idx += 256) {
    int i = idx >> 6, j = idx & 63;
    Tb[i][j] = (bf16)Tm[i][j];
  }
  __syncthreads();

  // v_c = T @ vb ; k_cumdecay = T @ (kb*exp(g))
  const long obase = ((long)(b * HV_ + hv) * NC_ + ic) * 64;
  for (int tt = wid; tt < 32; tt += 8) {
    int mt = tt >> 3, nt = tt & 7;
    v8f c0 = zero8(), c1 = zero8();
    for (int kk = 0; kk < 64; kk += 32) {
      v16bf a  = fragA(&Tb[0][0], 72, mt * 16, kk, lane);
      v16bf b0 = fragB(&vbT[0][0], 72, nt * 16, kk, lane);
      v16bf b1 = fragB(&kgT[0][0], 72, nt * 16, kk, lane);
      c0 = wmma_bf16(a, b0, c0);
      c1 = wmma_bf16(a, b1, c1);
    }
#pragma unroll
    for (int r = 0; r < 8; r++) {
      int i = mt * 16 + r + ((lane >> 4) << 3);
      int d = nt * 16 + (lane & 15);
      vc_h [(obase + i) * HD_ + d] = (bf16)c0[r];
      kcd_h[(obase + i) * HD_ + d] = (bf16)c1[r];
    }
  }
}

// ---------------- sequential chunk scan (block per (b,hv)) ----------------
__global__ void __launch_bounds__(256) chunk_scan(
    const bf16* __restrict__ qn_h, const bf16* __restrict__ kn_h,
    const bf16* __restrict__ vc_h, const bf16* __restrict__ kcd_h,
    const float* __restrict__ cumg_g, float* __restrict__ hcore) {
  __shared__ float stateF[128][129];   // state[d][e] master, f32
  __shared__ bf16  stateT[128][136];   // stateT[e][d] : B-operand (lane n=e, K=d contiguous)
  __shared__ bf16  knT[128][72];       // knT[d][t] = kn[t][d]*exp(glast-g_t)
  __shared__ bf16  vnT[128][72];       // v_new^T [e][t]
  __shared__ bf16  attnS[64][72];      // attn[t][s] masked+decayed
  __shared__ float cg[64];

  const int tid = threadIdx.x, lane = tid & 31, wid = tid >> 5;
  const int hv = blockIdx.x & 31, b = blockIdx.x >> 5;
  const int h = hv >> 1;
  const int qstride = HQK_ * HD_;      // halves between consecutive t rows

  for (int idx = tid; idx < 128 * 128; idx += 256) {
    int d = idx >> 7, e = idx & 127;
    stateF[d][e] = 0.f;
    stateT[e][d] = (bf16)0.f;
  }
  __syncthreads();

  for (int ic = 0; ic < NC_; ic++) {
    const int s0 = ic * CHUNK_;
    const long qbase = ((long)(b * S_ + s0) * HQK_ + h) * HD_;
    const long cbase = ((long)(b * HV_ + hv) * NC_ + ic) * 64;
    if (tid < 64) cg[tid] = cumg_g[cbase + tid];
    __syncthreads();

    // Phase A: attn = (qn @ kn^T) * decay (tril incl diag); build knT
    for (int tt = wid; tt < 16; tt += 8) {
      int mt = tt >> 2, nt = tt & 3;
      v8f c = zero8();
      for (int kk = 0; kk < 128; kk += 32) {
        v16bf a  = fragA(qn_h + qbase, qstride, mt * 16, kk, lane);
        v16bf bb = fragB(kn_h + qbase, qstride, nt * 16, kk, lane);
        c = wmma_bf16(a, bb, c);
      }
#pragma unroll
      for (int r = 0; r < 8; r++) {
        int t = mt * 16 + r + ((lane >> 4) << 3);
        int s = nt * 16 + (lane & 15);
        attnS[t][s] = (bf16)((s <= t) ? c[r] * expf(cg[t] - cg[s]) : 0.f);
      }
    }
    {
      float glast = cg[63];
      for (int idx = tid; idx < 64 * 128; idx += 256) {
        int t = idx >> 7, d = idx & 127;
        float kv = (float)kn_h[qbase + (long)t * qstride + d];
        knT[d][t] = (bf16)(kv * expf(glast - cg[t]));
      }
    }
    __syncthreads();

    // Phase B: v_new = v_c - kcd @ state   (stored transposed, bf16)
    for (int tt = wid; tt < 32; tt += 8) {
      int mt = tt >> 3, nt = tt & 7;
      v8f c = zero8();
      for (int kk = 0; kk < 128; kk += 32) {
        v16bf a  = fragA(kcd_h + cbase * HD_, HD_, mt * 16, kk, lane);
        v16bf bb = fragB(&stateT[0][0], 136, nt * 16, kk, lane);
        c = wmma_bf16(a, bb, c);
      }
#pragma unroll
      for (int r = 0; r < 8; r++) {
        int t = mt * 16 + r + ((lane >> 4) << 3);
        int e = nt * 16 + (lane & 15);
        float vcv = (float)vc_h[(cbase + t) * HD_ + e];
        vnT[e][t] = (bf16)(vcv - c[r]);
      }
    }
    __syncthreads();

    // Phase C: out = exp(g)_row * (qn @ state) + attn @ v_new
    for (int tt = wid; tt < 32; tt += 8) {
      int mt = tt >> 3, nt = tt & 7;
      v8f c = zero8();
      for (int kk = 0; kk < 128; kk += 32) {
        v16bf a  = fragA(qn_h + qbase, qstride, mt * 16, kk, lane);
        v16bf bb = fragB(&stateT[0][0], 136, nt * 16, kk, lane);
        c = wmma_bf16(a, bb, c);
      }
#pragma unroll
      for (int r = 0; r < 8; r++) {
        int t = mt * 16 + r + ((lane >> 4) << 3);
        c[r] *= expf(cg[t]);                 // per-row scale on C fragment
      }
      for (int kk = 0; kk < 64; kk += 32) {
        v16bf a  = fragA(&attnS[0][0], 72, mt * 16, kk, lane);
        v16bf bb = fragB(&vnT[0][0],  72, nt * 16, kk, lane);
        c = wmma_bf16(a, bb, c);
      }
#pragma unroll
      for (int r = 0; r < 8; r++) {
        int t = mt * 16 + r + ((lane >> 4) << 3);
        int d = nt * 16 + (lane & 15);
        hcore[((long)(b * S_ + s0 + t) * HV_ + hv) * HD_ + d] = c[r];
      }
    }
    __syncthreads();

    // Phase D: state = state*exp(glast) + knS^T @ v_new
    {
      float eglast = expf(cg[63]);
      for (int tt = wid; tt < 64; tt += 8) {
        int mt = tt >> 3, nt = tt & 7;
        v8f c = zero8();
        for (int kk = 0; kk < 64; kk += 32) {
          v16bf a  = fragA(&knT[0][0], 72, mt * 16, kk, lane);
          v16bf bb = fragB(&vnT[0][0], 72, nt * 16, kk, lane);
          c = wmma_bf16(a, bb, c);
        }
#pragma unroll
        for (int r = 0; r < 8; r++) {
          int d = mt * 16 + r + ((lane >> 4) << 3);
          int e = nt * 16 + (lane & 15);
          float ns = stateF[d][e] * eglast + c[r];
          stateF[d][e] = ns;
          stateT[e][d] = (bf16)ns;
        }
      }
    }
    __syncthreads();
  }
}

// ---------------- final gating: rmsnorm(h) * norm_w * silu(z) -> bf16 ----------------
__global__ void __launch_bounds__(256) gate_out(const float* __restrict__ hcore,
                                                const float* __restrict__ mqkvz,
                                                const float* __restrict__ norm_w,
                                                bf16* __restrict__ hg) {
  int lane = threadIdx.x & 31, wid = threadIdx.x >> 5;
  int row = blockIdx.x * 8 + wid;          // (b*S+s)*HV + hv
  if (row >= BS_ * HV_) return;
  int hv = row & 31; long bs = row >> 5;
  const float* p = hcore + (long)row * HD_;
  float v[4]; float ss = 0.f;
#pragma unroll
  for (int j = 0; j < 4; j++) { v[j] = p[lane + 32*j]; ss += v[j]*v[j]; }
#pragma unroll
  for (int m = 16; m > 0; m >>= 1) ss += __shfl_xor(ss, m, 32);
  float r = rsqrtf(ss * (1.f / HD_) + 1e-6f);
  long zbase = bs * OQ_ + (long)(hv >> 1) * 768 + 512 + (hv & 1) * 128;
#pragma unroll
  for (int j = 0; j < 4; j++) {
    int d = lane + 32*j;
    float z = mqkvz[zbase + d];
    float sz = z / (1.f + expf(-z));
    hg[(long)row * HD_ + d] = (bf16)(v[j] * r * norm_w[d] * sz);
  }
}

// ---------------- host launcher ----------------
extern "C" void kernel_launch(void* const* d_in, const int* in_sizes, int n_in,
                              void* d_out, int out_size, void* d_ws, size_t ws_size,
                              hipStream_t stream) {
  (void)in_sizes; (void)n_in; (void)out_size; (void)ws_size;
  const float* x       = (const float*)d_in[0];
  const float* W_qkvz  = (const float*)d_in[1];
  const float* W_ba    = (const float*)d_in[2];
  const float* conv_w  = (const float*)d_in[3];
  const float* dt_bias = (const float*)d_in[4];
  const float* A_log   = (const float*)d_in[5];
  const float* norm_w  = (const float*)d_in[6];
  const float* W_out   = (const float*)d_in[7];
  float* out = (float*)d_out;

  char* ws = (char*)d_ws;
  size_t off = 0;
  auto alloc = [&](size_t bytes) -> void* {
    void* p = ws + off;
    off += (bytes + 255) & ~(size_t)255;
    return p;
  };
  bf16*  xh     = (bf16*) alloc((size_t)BS_ * D_ * 2);
  bf16*  Wqh    = (bf16*) alloc((size_t)OQ_ * D_ * 2);
  bf16*  Wbah   = (bf16*) alloc((size_t)OBA_ * D_ * 2);
  bf16*  Wouth  = (bf16*) alloc((size_t)D_ * HV_ * HD_ * 2);
  float* mqkvz  = (float*)alloc((size_t)BS_ * OQ_ * 4);
  float* mba    = (float*)alloc((size_t)BS_ * OBA_ * 4);
  float* qconv  = (float*)alloc((size_t)BS_ * HQK_ * HD_ * 4);
  float* kconv  = (float*)alloc((size_t)BS_ * HQK_ * HD_ * 4);
  bf16*  vconvh = (bf16*) alloc((size_t)BS_ * HV_ * HD_ * 2);
  bf16*  qnh    = (bf16*) alloc((size_t)BS_ * HQK_ * HD_ * 2);
  bf16*  knh    = (bf16*) alloc((size_t)BS_ * HQK_ * HD_ * 2);
  float* gbuf   = (float*)alloc((size_t)BS_ * HV_ * 4);
  float* bbuf   = (float*)alloc((size_t)BS_ * HV_ * 4);
  float* cumg   = (float*)alloc((size_t)B_ * HV_ * NC_ * 64 * 4);
  bf16*  vch    = (bf16*) alloc((size_t)B_ * HV_ * NC_ * 64 * HD_ * 2);
  bf16*  kcdh   = (bf16*) alloc((size_t)B_ * HV_ * NC_ * 64 * HD_ * 2);
  float* hcore  = (float*)alloc((size_t)BS_ * HV_ * HD_ * 4);
  bf16*  hgh    = (bf16*) alloc((size_t)BS_ * HV_ * HD_ * 2);

  // 1) casts to bf16
  cast_f32_bf16<<<2048, 256, 0, stream>>>(x, xh, (long)BS_ * D_);
  cast_f32_bf16<<<4096, 256, 0, stream>>>(W_qkvz, Wqh, (long)OQ_ * D_);
  cast_f32_bf16<<<512,  256, 0, stream>>>(W_ba, Wbah, (long)OBA_ * D_);
  cast_f32_bf16<<<2048, 256, 0, stream>>>(W_out, Wouth, (long)D_ * HV_ * HD_);

  // 2) input projections (WMMA)
  { dim3 g(OQ_ / 128, BS_ / 128);
    gemm_nt_bf16<<<g, 256, 0, stream>>>(xh, Wqh, mqkvz, BS_, OQ_, D_); }
  { dim3 g(1, BS_ / 128);
    gemm_nt_bf16<<<g, 256, 0, stream>>>(xh, Wbah, mba, BS_, OBA_, D_); }

  // 3) conv+silu, gates, rmsnorms
  { dim3 g(CD_ / 256, B_);
    conv_silu_kernel<<<g, 256, 0, stream>>>(mqkvz, conv_w, qconv, kconv, vconvh); }
  gate_kernel<<<(BS_ * HV_ + 255) / 256, 256, 0, stream>>>(mba, A_log, dt_bias, gbuf, bbuf);
  rmsnorm_rows<<<BS_ * HQK_ / 8, 256, 0, stream>>>(qconv, qnh, 1.f / 128.f, BS_ * HQK_);
  rmsnorm_rows<<<BS_ * HQK_ / 8, 256, 0, stream>>>(kconv, knh, 0.08838834764831845f, BS_ * HQK_);

  // 4) chunked delta-rule core (WMMA)
  chunk_precompute<<<B_ * HV_ * NC_, 256, 0, stream>>>(knh, vconvh, gbuf, bbuf, cumg, vch, kcdh);
  chunk_scan<<<B_ * HV_, 256, 0, stream>>>(qnh, knh, vch, kcdh, cumg, hcore);

  // 5) gate + output projection (WMMA)
  gate_out<<<BS_ * HV_ / 8, 256, 0, stream>>>(hcore, mqkvz, norm_w, hgh);
  { dim3 g(D_ / 128, BS_ / 128);
    gemm_nt_bf16<<<g, 256, 0, stream>>>(hgh, Wouth, out, BS_, D_, HV_ * HD_); }
}
